// ConvSelfAttention_64957085384894
// MI455X (gfx1250) — matrix operand order repaired and checked by reading, not compiled, not run on hardware
//
#include <hip/hip_runtime.h>

// ---------------------------------------------------------------------------
// ConvSelfAttention for MI455X (gfx1250): bf16 WMMA everywhere, f32 accumulate
// Round 3: 32x64 register blocking (8 WMMA chains/wave), B-fragment reuse
// across two M-subtiles, near-cache prefetch, banded WMMA attention.
// ---------------------------------------------------------------------------

typedef __bf16 bf16_t;
typedef __attribute__((ext_vector_type(16))) __bf16 v16bf;
typedef __attribute__((ext_vector_type(8)))  __bf16 v8bf;
typedef __attribute__((ext_vector_type(8)))  float   v8f;

#define EMB    128
#define HEADS  8
#define KWIN   32
#define BATCH  2
#define TLEN   2048
#define TPAD   (TLEN + KWIN - 1)   // 2079
#define HE     (HEADS * EMB)       // 1024

#define WMMA_BF16(a, b, c) \
    __builtin_amdgcn_wmma_f32_16x16x32_bf16(false, (a), false, (b), (short)0, (c), false, false)

// ---- WMMA fragment loaders (ISA 7.12.2 lane layouts) -----------------------
// A 16x32 bf16: lane m=l&15, half h=l>>4; VGPR0-3 = K 8h..8h+7, VGPR4-7 = K 16+8h..+7.
static __device__ __forceinline__ v16bf load_a_frag(const bf16_t* row_k0, int khalf) {
    v16bf a;
    ((v8bf*)&a)[0] = *(const v8bf*)(row_k0 + 8 * khalf);
    ((v8bf*)&a)[1] = *(const v8bf*)(row_k0 + 16 + 8 * khalf);
    return a;
}
// B 32x16 bf16: lane n=l&15, half h=l>>4 holds K=8h..8h+7 and 16+8h..+7 of column n
// (column stored contiguously = weight row / K^T row).
static __device__ __forceinline__ v16bf load_b_frag(const bf16_t* col_k0, int khalf) {
    v16bf b;
    ((v8bf*)&b)[0] = *(const v8bf*)(col_k0 + 8 * khalf);
    ((v8bf*)&b)[1] = *(const v8bf*)(col_k0 + 16 + 8 * khalf);
    return b;
}

// ---- elementwise convert / pad --------------------------------------------
__global__ void cvt_f32_to_bf16(const float* __restrict__ s, bf16_t* __restrict__ d, int n) {
    int i = blockIdx.x * blockDim.x + threadIdx.x;
    if (i < n) d[i] = (bf16_t)s[i];
}

__global__ void build_xpad_bf16(const float* __restrict__ x, bf16_t* __restrict__ xp) {
    int i = blockIdx.x * blockDim.x + threadIdx.x;
    const int n = BATCH * TPAD * EMB;
    if (i >= n) return;
    int e   = i % EMB;
    int row = (i / EMB) % TPAD;
    int b   = i / (EMB * TPAD);
    float v = 0.0f;
    if (row >= KWIN - 1)
        v = x[((size_t)b * TLEN + (row - (KWIN - 1))) * EMB + e];
    xp[i] = (bf16_t)v;
}

// ---- bf16 WMMA GEMM: D[M][N] = A[M][K] * W[N][K]^T (+bias) -----------------
// One wave per 32x64 tile: 2 M-subtiles x 4 N-subtiles = 8 accumulator chains.
// B fragments amortized over both M-subtiles; A fragments over 4 N-subtiles.
// Block = 256 threads = 8 waves; grid.y covers N in steps of 64.
__global__ void gemm_bf16_wmma(const bf16_t* __restrict__ A,
                               const bf16_t* __restrict__ W,
                               bf16_t* __restrict__ Dbf,    // bf16 out (or null)
                               float*  __restrict__ Df,     // f32 out (or null)
                               const float* __restrict__ bias,
                               int M, int N, int K)
{
    const int lane  = threadIdx.x & 31;
    const int wave  = threadIdx.x >> 5;
    const int WAVES = blockDim.x >> 5;
    const int MT    = (M + 31) >> 5;          // 32-row tiles
    const int mtile = blockIdx.x * WAVES + wave;
    if (mtile >= MT) return;                  // wave-uniform: EXEC stays all-1s
    const int m0 = mtile << 5;
    const int n0 = (int)blockIdx.y << 6;      // 64-wide N block
    const int khalf = lane >> 4;
    const int l15   = lane & 15;

    int ar0 = m0 + l15;      if (ar0 > M - 1) ar0 = M - 1;   // clamp tail rows
    int ar1 = m0 + 16 + l15; if (ar1 > M - 1) ar1 = M - 1;
    const bf16_t* arow0 = A + (size_t)ar0 * K;
    const bf16_t* arow1 = A + (size_t)ar1 * K;
    const bf16_t* wrow  = W + (size_t)(n0 + l15) * K;        // subtile j adds 16*K

    v8f c00 = {}, c01 = {}, c02 = {}, c03 = {};
    v8f c10 = {}, c11 = {}, c12 = {}, c13 = {};
#pragma unroll 2
    for (int k0 = 0; k0 < K; k0 += 32) {
        if (k0 + 64 < K) {
            __builtin_prefetch(arow0 + k0 + 64, 0, 3);       // global_prefetch_b8
            __builtin_prefetch(arow1 + k0 + 64, 0, 3);
        }
        v16bf a0 = load_a_frag(arow0 + k0, khalf);
        v16bf a1 = load_a_frag(arow1 + k0, khalf);
        v16bf b0 = load_b_frag(wrow + k0, khalf);
        v16bf b1 = load_b_frag(wrow + (size_t)16 * K + k0, khalf);
        v16bf b2 = load_b_frag(wrow + (size_t)32 * K + k0, khalf);
        v16bf b3 = load_b_frag(wrow + (size_t)48 * K + k0, khalf);
        c00 = WMMA_BF16(a0, b0, c00);
        c10 = WMMA_BF16(a1, b0, c10);
        c01 = WMMA_BF16(a0, b1, c01);
        c11 = WMMA_BF16(a1, b1, c11);
        c02 = WMMA_BF16(a0, b2, c02);
        c12 = WMMA_BF16(a1, b2, c12);
        c03 = WMMA_BF16(a0, b3, c03);
        c13 = WMMA_BF16(a1, b3, c13);
    }

    // D layout (ISA 7.12.2): lane n = l&15, rows m = 8*(l>>4)+r in VGPR r.
    const v8f* accs[2][4] = { { &c00, &c01, &c02, &c03 },
                              { &c10, &c11, &c12, &c13 } };
#pragma unroll
    for (int half = 0; half < 2; ++half) {
#pragma unroll
        for (int j = 0; j < 4; ++j) {
            const int n = n0 + 16 * j + l15;
            v8f a = *accs[half][j];
#pragma unroll
            for (int r = 0; r < 8; ++r) {
                int m = m0 + half * 16 + khalf * 8 + r;
                if (m < M) {
                    if (Dbf) Dbf[(size_t)m * N + n] = (bf16_t)a[r];
                    else     Df [(size_t)m * N + n] = a[r] + (bias ? bias[n] : 0.0f);
                }
            }
        }
    }
}

// ---- banded attention: 1 wave per (b, head, 16-row t-tile) -----------------
__global__ void attn_banded_wmma(const bf16_t* __restrict__ Q,   // [B,T,HE]
                                 const bf16_t* __restrict__ Kb,  // [B,TPAD,HE]
                                 const bf16_t* __restrict__ Vb,  // [B,TPAD,HE]
                                 bf16_t* __restrict__ O)         // [B,T,HE]
{
    __shared__ float  Sld[16][48];        // raw score band panel
    __shared__ bf16_t Pld[16 * 64];       // probabilities (K-dim padded to 64)
    __shared__ bf16_t VT [128 * 64];      // V transposed: [e][s]

    const int lane  = threadIdx.x & 31;
    const int khalf = lane >> 4;
    const int l15   = lane & 15;
    const int t0    = (int)blockIdx.x << 4;
    const int b     = (int)blockIdx.y / HEADS;
    const int head  = (int)blockIdx.y % HEADS;

    const size_t qbase = (size_t)b * TLEN * HE + (size_t)head * EMB;
    const size_t kbase = (size_t)b * TPAD * HE + (size_t)head * EMB;

    // Stage V^T tile: columns c=0..46 are padded positions t0+c; rest zero.
    for (int idx = lane; idx < 128 * 64; idx += 32) {
        int c = idx >> 7, e = idx & 127;          // e fastest -> coalesced reads
        int s = t0 + c;
        bf16_t v = (bf16_t)0.0f;
        if (c < KWIN + 15 && s < TPAD)
            v = Vb[kbase + (size_t)s * HE + e];
        VT[e * 64 + c] = v;
    }

    // Scores S[m][c] = q[t0+m].k[t0+c], c in [0,48): Q fragment shared by 3 N-tiles.
    const bf16_t* qrow = Q + qbase + (size_t)(t0 + l15) * HE;
    int s2 = t0 + 32 + l15; if (s2 > TPAD - 1) s2 = TPAD - 1;    // c=47 masked later
    const bf16_t* kr0 = Kb + kbase + (size_t)(t0 + l15) * HE;
    const bf16_t* kr1 = Kb + kbase + (size_t)(t0 + 16 + l15) * HE;
    const bf16_t* kr2 = Kb + kbase + (size_t)s2 * HE;
    v8f sc0 = {}, sc1 = {}, sc2 = {};
#pragma unroll
    for (int e0 = 0; e0 < EMB; e0 += 32) {
        v16bf a = load_a_frag(qrow + e0, khalf);
        sc0 = WMMA_BF16(a, load_b_frag(kr0 + e0, khalf), sc0);
        sc1 = WMMA_BF16(a, load_b_frag(kr1 + e0, khalf), sc1);
        sc2 = WMMA_BF16(a, load_b_frag(kr2 + e0, khalf), sc2);
    }
#pragma unroll
    for (int r = 0; r < 8; ++r) {
        Sld[khalf * 8 + r][l15]      = sc0[r];
        Sld[khalf * 8 + r][16 + l15] = sc1[r];
        Sld[khalf * 8 + r][32 + l15] = sc2[r];
    }
    __syncthreads();

    // Masked softmax: row m attends to c in [m, m+31]. Lanes 0..15, one row each.
    if (lane < 16) {
        const int m = lane;
        const float scale = 0.08838834764831845f;  // 1/sqrt(EMB) (q,k each /e^0.25)
        float mx = -3.0e38f;
        for (int j = 0; j < KWIN; ++j)
            mx = fmaxf(mx, Sld[m][m + j] * scale);
        float sum = 0.0f;
        for (int j = 0; j < KWIN; ++j)
            sum += __expf(Sld[m][m + j] * scale - mx);
        float inv = 1.0f / sum;
        for (int c = 0; c < 64; ++c) {
            float p = 0.0f;
            int j = c - m;
            if (j >= 0 && j < KWIN)
                p = __expf(Sld[m][c] * scale - mx) * inv;
            Pld[m * 64 + c] = (bf16_t)p;
        }
    }
    __syncthreads();

    // Out[m][e] = sum_c P[m][c]*V[c][e]: P fragments hoisted, shared by 8 N-tiles.
    v16bf pa0 = load_a_frag(Pld + l15 * 64, khalf);       // ds_load_b128 x2
    v16bf pa1 = load_a_frag(Pld + l15 * 64 + 32, khalf);
#pragma unroll
    for (int et = 0; et < 8; ++et) {
        const bf16_t* vtr = VT + (size_t)(et * 16 + l15) * 64;
        v8f acc = {};
        acc = WMMA_BF16(pa0, load_b_frag(vtr, khalf), acc);
        acc = WMMA_BF16(pa1, load_b_frag(vtr + 32, khalf), acc);
#pragma unroll
        for (int r = 0; r < 8; ++r)
            O[qbase + (size_t)(t0 + khalf * 8 + r) * HE + et * 16 + l15] = (bf16_t)acc[r];
    }
}

// ---------------------------------------------------------------------------
extern "C" void kernel_launch(void* const* d_in, const int* in_sizes, int n_in,
                              void* d_out, int out_size, void* d_ws, size_t ws_size,
                              hipStream_t stream)
{
    (void)in_sizes; (void)n_in; (void)out_size; (void)ws_size;
    const float* x  = (const float*)d_in[0];
    const float* Wq = (const float*)d_in[1];
    const float* Wk = (const float*)d_in[2];
    const float* Wv = (const float*)d_in[3];
    const float* Wu = (const float*)d_in[4];
    const float* bu = (const float*)d_in[5];
    float* out = (float*)d_out;

    char* ws = (char*)d_ws;
    size_t off = 0;
    auto alloc = [&](size_t bytes) -> void* {
        void* p = ws + off;
        off = (off + bytes + 255) & ~(size_t)255;
        return p;
    };
    bf16_t* xbf  = (bf16_t*)alloc((size_t)BATCH * TLEN * EMB * 2);
    bf16_t* xpbf = (bf16_t*)alloc((size_t)BATCH * TPAD * EMB * 2);
    bf16_t* wqbf = (bf16_t*)alloc((size_t)HE * EMB * 2);
    bf16_t* wkbf = (bf16_t*)alloc((size_t)HE * EMB * 2);
    bf16_t* wvbf = (bf16_t*)alloc((size_t)HE * EMB * 2);
    bf16_t* wubf = (bf16_t*)alloc((size_t)EMB * HE * 2);
    bf16_t* qbf  = (bf16_t*)alloc((size_t)BATCH * TLEN * HE * 2);
    bf16_t* kbf  = (bf16_t*)alloc((size_t)BATCH * TPAD * HE * 2);
    bf16_t* vbf  = (bf16_t*)alloc((size_t)BATCH * TPAD * HE * 2);
    bf16_t* obf  = (bf16_t*)alloc((size_t)BATCH * TLEN * HE * 2);

    // 1) convert / pad to bf16
    {
        int n = BATCH * TLEN * EMB;
        cvt_f32_to_bf16<<<(n + 255) / 256, 256, 0, stream>>>(x, xbf, n);
        int np = BATCH * TPAD * EMB;
        build_xpad_bf16<<<(np + 255) / 256, 256, 0, stream>>>(x, xpbf);
        int nw = HE * EMB;
        cvt_f32_to_bf16<<<(nw + 255) / 256, 256, 0, stream>>>(Wq, wqbf, nw);
        cvt_f32_to_bf16<<<(nw + 255) / 256, 256, 0, stream>>>(Wk, wkbf, nw);
        cvt_f32_to_bf16<<<(nw + 255) / 256, 256, 0, stream>>>(Wv, wvbf, nw);
        cvt_f32_to_bf16<<<(nw + 255) / 256, 256, 0, stream>>>(Wu, wubf, nw);
    }

    // 2) Q/K/V projections (bf16 WMMA, f32 accumulate, bf16 intermediates)
    {
        dim3 blk(256);
        int Mq = BATCH * TLEN, Mk = BATCH * TPAD;
        int MTq = (Mq + 31) / 32, MTk = (Mk + 31) / 32;
        dim3 gq((MTq + 7) / 8, HE / 64), gk((MTk + 7) / 8, HE / 64);
        gemm_bf16_wmma<<<gq, blk, 0, stream>>>(xbf,  wqbf, qbf, nullptr, nullptr, Mq, HE, EMB);
        gemm_bf16_wmma<<<gk, blk, 0, stream>>>(xpbf, wkbf, kbf, nullptr, nullptr, Mk, HE, EMB);
        gemm_bf16_wmma<<<gk, blk, 0, stream>>>(xpbf, wvbf, vbf, nullptr, nullptr, Mk, HE, EMB);
    }

    // 3) banded attention
    {
        dim3 ga(TLEN / 16, BATCH * HEADS);
        attn_banded_wmma<<<ga, 32, 0, stream>>>(qbf, kbf, vbf, obf);
    }

    // 4) output projection + bias -> f32 d_out
    {
        dim3 blk(256);
        int M = BATCH * TLEN;
        dim3 g(((M + 31) / 32 + 7) / 8, EMB / 64);
        gemm_bf16_wmma<<<g, blk, 0, stream>>>(obf, wubf, nullptr, out, bu, M, EMB, HE);
    }
}
